// GraphAttentionLayer_83013127897467
// MI455X (gfx1250) — compile-verified, matrix operand order checked by reading, not run on hardware
//
#include <hip/hip_runtime.h>
#include <hip/hip_bf16.h>
#include <stdint.h>

typedef __attribute__((ext_vector_type(16))) _Float16 v16h;
typedef __attribute__((ext_vector_type(8)))  float    v8f;

#define T_DIM   128
#define N_NODES 38
#define F_OUT   256
#define M_PAD   48
#define LRELU_A 0.2f

// D = A*B + C, f16 inputs, f32 accumulate (V_WMMA_F32_16X16X32_F16)
__device__ __forceinline__ v8f wmma_f16(v16h a, v16h b, v8f c) {
  return __builtin_amdgcn_wmma_f32_16x16x32_f16(false, a, false, b, (short)0, c,
                                                false, false);
}

// Fragment loader. Data stored as u32 pairs of f16 along K:
//   base[row * strideU32 + kp],  kp = K/2.
// 16-bit A-matrix layout (ISA 7.12.2): lanes 0-15 hold K 0-7 (VGPR0-3) and
// K 16-23 (VGPR4-7); lanes 16-31 hold K 8-15 and K 24-31. B assumed mirrored
// over N (row = N column for B).
__device__ __forceinline__ v16h load_frag(const uint32_t* base, int row,
                                          int strideU32, int kpbase, int lane) {
  union { v16h h; uint32_t u[8]; } f;
  const int half = (lane >> 4) & 1;
  const uint32_t* p = base + row * strideU32 + kpbase + half * 4;
#pragma unroll
  for (int v = 0; v < 4; ++v) f.u[v] = p[v];
#pragma unroll
  for (int v = 0; v < 4; ++v) f.u[4 + v] = p[8 + v];
  return f.h;
}

// One-time prep: pack W -> f16 pairs (B-matrix layout Wp[f][t/2]) and build
// the cosine-similarity top-k adjacency mask from emb.
__global__ void gat_prep_kernel(const float* __restrict__ W,
                                const float* __restrict__ emb,
                                uint32_t* __restrict__ Wp,
                                uint32_t* __restrict__ mask) {
  __shared__ float adjS[N_NODES * N_NODES];
  __shared__ float normS[N_NODES];
  __shared__ float thrS[N_NODES];
  const int tid = threadIdx.x;

  for (int idx = tid; idx < F_OUT * (T_DIM / 2); idx += 256) {
    int n = idx >> 6;
    int kp = idx & 63;
    union { _Float16 h[2]; uint32_t u; } pk;
    pk.h[0] = (_Float16)W[(2 * kp) * F_OUT + n];
    pk.h[1] = (_Float16)W[(2 * kp + 1) * F_OUT + n];
    Wp[idx] = pk.u;
  }

  if (tid < N_NODES) {
    float s = 0.f;
    for (int k = 0; k < 128; ++k) { float v = emb[tid * 128 + k]; s += v * v; }
    normS[tid] = sqrtf(s);
  }
  __syncthreads();
  for (int idx = tid; idx < N_NODES * N_NODES; idx += 256) {
    int i = idx / N_NODES, j = idx - i * N_NODES;
    float s = 0.f;
    for (int k = 0; k < 128; ++k) s += emb[i * 128 + k] * emb[j * 128 + k];
    adjS[idx] = s / (normS[i] * normS[j]);
  }
  __syncthreads();
  if (tid < N_NODES) {
    float row[N_NODES];
    for (int j = 0; j < N_NODES; ++j) row[j] = adjS[tid * N_NODES + j];
    // partial selection sort: row[8] = 9th largest (top_k(.,10)[...,-2])
    for (int s = 0; s < 9; ++s) {
      int am = s;
      for (int j = s + 1; j < N_NODES; ++j) if (row[j] > row[am]) am = j;
      float t = row[s]; row[s] = row[am]; row[am] = t;
    }
    thrS[tid] = row[8];
  }
  __syncthreads();
  for (int idx = tid; idx < N_NODES * N_NODES; idx += 256) {
    int i = idx / N_NODES;
    float v = adjS[idx];
    mask[idx] = ((v > thrS[i]) || (v == 1.0f)) ? 1u : 0u;
  }
}

// One block per batch element; 8 waves (wave32).
__global__ void __launch_bounds__(256)
gat_main_kernel(const float* __restrict__ x, const float* __restrict__ a,
                const uint32_t* __restrict__ Wp, const uint32_t* __restrict__ mask,
                float* __restrict__ out) {
  // Aliased LDS regions (total 63808 B):
  //  [0,30720):      xA (phase A-B) | attA (0..6144) + outS (6144..30720) later
  //  [30720,63488):  xraw (phase A) | hB later
  //  [63488,63808):  slS / srS
  __shared__ __align__(16) unsigned char smem[63808];
  uint32_t* xA   = (uint32_t*)(smem);          // 48*64  u32
  uint32_t* attA = (uint32_t*)(smem);          // 48*32  u32
  float*    outS = (float*)(smem + 6144);      // 128*48 f32
  float*    xraw = (float*)(smem + 30720);     // 128*38 f32
  uint32_t* hB   = (uint32_t*)(smem + 30720);  // 256*32 u32
  float*    slS  = (float*)(smem + 63488);
  float*    srS  = (float*)(smem + 63648);

  const int tid  = threadIdx.x;
  const int lane = tid & 31;
  const int wave = tid >> 5;
  const int half = (lane >> 4) & 1;
  const int b    = blockIdx.x;

  // ---- Phase A: async direct-to-LDS stage of x[b] (128x38 f32, 4864 dwords;
  //      exactly 19 full iterations of 256 lanes). GLOBAL_LOAD_ASYNC_TO_LDS
  //      bypasses the VGPR round-trip; tracked by ASYNCcnt. ----
  {
    const float* src = x + (size_t)b * (T_DIM * N_NODES) + tid;
    uint32_t dst = (uint32_t)(uintptr_t)(xraw + tid);  // low 32 bits = LDS addr
#pragma unroll
    for (int it = 0; it < (T_DIM * N_NODES) / 256; ++it) {
      asm volatile("global_load_async_to_lds_b32 %0, %1, off"
                   :: "v"(dst), "v"(src)
                   : "memory");
      src += 256;
      dst += 256 * 4;
    }
    asm volatile("s_wait_asynccnt 0x0" ::: "memory");
  }
  __syncthreads();

  // ---- Phase A2: A = x^T as f16 pairs over K=t; rows 38..47 zero ----
  for (int idx = tid; idx < M_PAD * 64; idx += 256) {
    int m = idx >> 6, kp = idx & 63;
    union { _Float16 h[2]; uint32_t u; } pk;
    if (m < N_NODES) {
      pk.h[0] = (_Float16)xraw[(2 * kp) * N_NODES + m];
      pk.h[1] = (_Float16)xraw[(2 * kp + 1) * N_NODES + m];
    } else {
      pk.u = 0u;
    }
    xA[idx] = pk.u;
  }
  __syncthreads();

  // zero hB (aliases dead xraw); padded K rows 38..63 must be 0
  for (int i = tid; i < 256 * 32; i += 256) hB[i] = 0u;

  // ---- Phase B: GEMM1  h = x^T W   (M=48, N=256, K=128) ----
  const int nt0 = wave * 2, nt1 = wave * 2 + 1;
  const int rA = lane & 15;
  v8f zero8 = {0.f, 0.f, 0.f, 0.f, 0.f, 0.f, 0.f, 0.f};
  v8f acc[3][2];
#pragma unroll
  for (int mi = 0; mi < 3; ++mi) { acc[mi][0] = zero8; acc[mi][1] = zero8; }
#pragma unroll
  for (int kk = 0; kk < 4; ++kk) {
    const int kpbase = kk * 16;
    v16h bf0 = load_frag(Wp, nt0 * 16 + rA, 64, kpbase, lane);
    v16h bf1 = load_frag(Wp, nt1 * 16 + rA, 64, kpbase, lane);
#pragma unroll
    for (int mi = 0; mi < 3; ++mi) {
      v16h af = load_frag(xA, mi * 16 + rA, 64, kpbase, lane);
      acc[mi][0] = wmma_f16(af, bf0, acc[mi][0]);
      acc[mi][1] = wmma_f16(af, bf1, acc[mi][1]);
    }
  }
  __syncthreads();  // hB zeroed everywhere; xA reads finished

  // ---- Phase C: h -> hB f16 (B-matrix layout, K=j); zero attA ----
  _Float16* hB_h = (_Float16*)hB;
#pragma unroll
  for (int ti = 0; ti < 2; ++ti) {
    int n = (wave * 2 + ti) * 16 + rA;
#pragma unroll
    for (int mi = 0; mi < 3; ++mi) {
#pragma unroll
      for (int v = 0; v < 8; ++v) {
        int j = mi * 16 + half * 8 + v;           // row of h (node), < 48
        hB_h[n * 64 + j] = (_Float16)acc[mi][ti][v];
      }
    }
  }
  for (int i = tid; i < 48 * 32; i += 256) attA[i] = 0u;
  __syncthreads();

  // ---- Phase D: s_left / s_right ----
  if (tid < N_NODES) {
    float sl = 0.f, sr = 0.f;
    for (int n = 0; n < F_OUT; ++n) {
      float hv = (float)hB_h[n * 64 + tid];
      sl += hv * a[n];
      sr += hv * a[F_OUT + n];
    }
    slS[tid] = sl;
    srS[tid] = sr;
  }
  __syncthreads();

  // ---- Phase E: masked softmax over i (per column j); att as f16 A-matrix ----
  _Float16* attA_h = (_Float16*)attA;
  if (tid < N_NODES) {
    const int j = tid;
    const float srj = srS[j];
    float m = -3.0e38f;
    for (int i = 0; i < N_NODES; ++i) {
      if (mask[i * N_NODES + j]) {
        float e = slS[i] + srj;
        e = e > 0.f ? e : LRELU_A * e;
        if (e > m) m = e;
      }
    }
    float s = 0.f;
    for (int i = 0; i < N_NODES; ++i) {
      if (mask[i * N_NODES + j]) {
        float e = slS[i] + srj;
        e = e > 0.f ? e : LRELU_A * e;
        s += __expf(e - m);
      }
    }
    const float inv = 1.0f / s;
    for (int i = 0; i < N_NODES; ++i) {
      float att = 0.f;  // NEG_INF entries underflow to exactly 0 after softmax
      if (mask[i * N_NODES + j]) {
        float e = slS[i] + srj;
        e = e > 0.f ? e : LRELU_A * e;
        att = __expf(e - m) * inv;
      }
      attA_h[i * 64 + j] = (_Float16)att;
    }
  }
  __syncthreads();

  // ---- Phase F: GEMM2  h' = att @ h   (M=48, N=256, K=64) ----
  v8f acc2[3][2];
#pragma unroll
  for (int mi = 0; mi < 3; ++mi) { acc2[mi][0] = zero8; acc2[mi][1] = zero8; }
#pragma unroll
  for (int kk = 0; kk < 2; ++kk) {
    const int kpbase = kk * 16;
    v16h bf0 = load_frag(hB, nt0 * 16 + rA, 32, kpbase, lane);
    v16h bf1 = load_frag(hB, nt1 * 16 + rA, 32, kpbase, lane);
#pragma unroll
    for (int mi = 0; mi < 3; ++mi) {
      v16h af = load_frag(attA, mi * 16 + rA, 32, kpbase, lane);
      acc2[mi][0] = wmma_f16(af, bf0, acc2[mi][0]);
      acc2[mi][1] = wmma_f16(af, bf1, acc2[mi][1]);
    }
  }

  // ---- Phase G: elu + LDS transpose, coalesced stores; 2 feature passes ----
  float* outp = out + (size_t)b * (F_OUT * N_NODES);
#pragma unroll
  for (int p = 0; p < 2; ++p) {
    __syncthreads();
    if ((wave >> 2) == p) {  // waves 0-3 own f<128, waves 4-7 own f>=128
#pragma unroll
      for (int ti = 0; ti < 2; ++ti) {
        int fl = ((wave * 2 + ti) * 16 + rA) - p * 128;  // local feature
#pragma unroll
        for (int mi = 0; mi < 3; ++mi) {
#pragma unroll
          for (int v = 0; v < 8; ++v) {
            int i = mi * 16 + half * 8 + v;
            float hv = acc2[mi][ti][v];
            outS[fl * 48 + i] = hv > 0.f ? hv : (__expf(hv) - 1.0f);  // elu
          }
        }
      }
    }
    __syncthreads();
    for (int idx = tid; idx < 128 * N_NODES; idx += 256) {
      int fl = idx / N_NODES, i = idx - fl * N_NODES;
      outp[p * 128 * N_NODES + idx] = outS[fl * 48 + i];
    }
  }
}

extern "C" void kernel_launch(void* const* d_in, const int* in_sizes, int n_in,
                              void* d_out, int out_size, void* d_ws, size_t ws_size,
                              hipStream_t stream) {
  const float* x   = (const float*)d_in[0];  // [B,128,38]
  const float* W   = (const float*)d_in[1];  // [128,256]
  const float* a   = (const float*)d_in[2];  // [512]
  const float* emb = (const float*)d_in[3];  // [38,128]
  float* out = (float*)d_out;                // [B,256,38]

  uint32_t* Wp    = (uint32_t*)d_ws;                          // 256*64 u32 = 64KB
  uint32_t* maskp = (uint32_t*)((char*)d_ws + 256 * 64 * 4);  // 38*38 u32

  const int B = in_sizes[0] / (T_DIM * N_NODES);

  gat_prep_kernel<<<1, 256, 0, stream>>>(W, emb, Wp, maskp);
  gat_main_kernel<<<B, 256, 0, stream>>>(x, a, Wp, maskp, out);
}
